// ConvAttention_61778809585927
// MI455X (gfx1250) — matrix operand order repaired and checked by reading, not compiled, not run on hardware
//
#include <hip/hip_runtime.h>

typedef __attribute__((ext_vector_type(16))) _Float16 v16h;
typedef __attribute__((ext_vector_type(8)))  _Float16 v8h;
typedef __attribute__((ext_vector_type(8)))  float    v8f;
typedef __attribute__((ext_vector_type(4)))  float    v4f;

// Problem constants (from reference setup_inputs)
constexpr int CB   = 16;   // batch
constexpr int CIN  = 32;
constexpr int COUT = 64;
constexpr int CH   = 16;   // H
constexpr int CW   = 512;  // W
constexpr int NT   = 256;  // threads per block (8 wave32)
constexpr float LRELU  = 0.3f;
constexpr float BN_EPS = 1e-5f;

// ---- dynamic LDS layout (bytes) ----
constexpr int SM_QP  = 0;                     // [512][32] f16 = 32768  Q^T rows (h-padded)
constexpr int SM_KP  = SM_QP + CW*32*2;       // 32768                  K^T rows (h-padded)
constexpr int SM_V   = SM_KP + CW*32*2;       // [16][512] f16 = 16384
constexpr int SM_PE  = SM_V  + CH*CW*2;       // [16][512] f32 = 32768
// double-buffered async x staging (phase 1); stats+P scratch overlay buffer A (phases 2-3)
constexpr int SM_UN  = SM_PE + CH*CW*4;       // 114688
constexpr int SM_XA  = SM_UN;                 // 65536
constexpr int SM_XB  = SM_UN + 65536;         // 65536
constexpr int SM_M   = SM_UN;                 // 512 f32 (row max)        [overlay XA]
constexpr int SM_L   = SM_UN + 2048;          // 512 f32 (1/row-sum)      [overlay XA]
constexpr int SM_P   = SM_UN + 4096;          // 8 waves * 16*32 f16 = 8192 [overlay XA]
constexpr int SM_WGT = SM_UN + 131072;        // 4*160 f32 = 2560 scaled conv weights
constexpr int SM_SH  = SM_WGT + 4*CIN*5*4;    // 4 f32 BN shifts
constexpr int SM_TOTAL = SM_SH + 4*4;         // 248336 bytes (< 320KB WGP LDS)

// Issue one h-row block (32 cin x 512 w f32 = 64KB) HBM -> LDS via async DMA.
// Each of the 8 waves issues 16 b128 ops (512B each) => ASYNCcnt rises by 16/wave.
__device__ __forceinline__ void stage_x_async(char* ldsbuf, const float* xrow, int tid) {
#pragma unroll
  for (int k = 0; k < 16; ++k) {
    int chunk = k * NT + tid;                       // 0..4095 16-byte chunks
    int ci    = chunk >> 7;                         // 128 chunks per cin row
    int rem   = chunk & 127;
    unsigned voff = (unsigned)(ci * (CH * CW * 4) + rem * 16);   // global byte offset
    unsigned ldsa = (unsigned)(unsigned long long)(ldsbuf + chunk * 16); // LDS byte addr
    asm volatile("global_load_async_to_lds_b128 %0, %1, %2"
                 :: "v"(ldsa), "v"(voff), "s"(xrow) : "memory");
  }
}

__global__ __launch_bounds__(NT) void convattn_kernel(
    const float* __restrict__ x,
    const float* __restrict__ wq, const float* __restrict__ gq, const float* __restrict__ bq,
    const float* __restrict__ muq, const float* __restrict__ vaq,
    const float* __restrict__ wk, const float* __restrict__ gk, const float* __restrict__ bk_,
    const float* __restrict__ muk, const float* __restrict__ vak,
    const float* __restrict__ wv, const float* __restrict__ gv, const float* __restrict__ bv,
    const float* __restrict__ muv, const float* __restrict__ vav,
    const float* __restrict__ wp, const float* __restrict__ gp, const float* __restrict__ bp,
    const float* __restrict__ mup, const float* __restrict__ vap,
    float* __restrict__ out)
{
  extern __shared__ char smem[];
  _Float16* sQp = (_Float16*)(smem + SM_QP);
  _Float16* sKp = (_Float16*)(smem + SM_KP);
  _Float16* sV  = (_Float16*)(smem + SM_V);
  float*    sPE = (float*)   (smem + SM_PE);
  float*    sM  = (float*)   (smem + SM_M);
  float*    sL  = (float*)   (smem + SM_L);
  _Float16* sP  = (_Float16*)(smem + SM_P);
  float*    sWgt= (float*)   (smem + SM_WGT);
  float*    sSh = (float*)   (smem + SM_SH);
  char*     xbufA = smem + SM_XA;     // scalar pointers (no static-initialized array
  char*     xbufB = smem + SM_XB;     // of LDS-derived pointers -> avoids lld error)

  const int tid    = threadIdx.x;
  const int lane   = tid & 31;
  const int wave   = tid >> 5;
  const int lane16 = lane & 15;
  const int half   = lane >> 4;       // 0: lanes 0-15, 1: lanes 16-31

  const int bc = blockIdx.x;          // 0..1023
  const int b  = bc / COUT;
  const int c  = bc % COUT;

  const float* xbase = x + ((size_t)b * CIN * CH) * CW;   // &x[b][0][h][0] at +h*CW

  // Kick off the DMA for h=0 immediately (overlaps with phase 0 below).
  stage_x_async(xbufA, xbase, tid);

  const float* Wb[4] = {wq, wk, wv, wp};
  const float* Gb[4] = {gq, gk, gv, gp};
  const float* Bb[4] = {bq, bk_, bv, bp};
  const float* Mb[4] = {muq, muk, muv, mup};
  const float* Vb[4] = {vaq, vak, vav, vap};

  // ---------------- Phase 0: zero-fill padded Q/K + BN-folded weights ----------------
  for (int i = tid; i < CW * 32 / 2; i += NT) {   // 8192 dwords each
    ((unsigned int*)sQp)[i] = 0u;
    ((unsigned int*)sKp)[i] = 0u;
  }
  for (int i = tid; i < 4 * CIN * 5; i += NT) {
    int br = i / (CIN * 5);
    int r  = i - br * (CIN * 5);
    float sc = Gb[br][c] * rsqrtf(Vb[br][c] + BN_EPS);
    sWgt[i] = Wb[br][(size_t)c * CIN * 5 + r] * sc;
  }
  if (tid < 4) {
    float sc = Gb[tid][c] * rsqrtf(Vb[tid][c] + BN_EPS);
    sSh[tid] = Bb[tid][c] - Mb[tid][c] * sc;
  }
  __syncthreads();

  // ---------------- Phase 1: conv(1x5) + BN + LeakyReLU, double-buffered DMA ----------
  for (int h = 0; h < CH; ++h) {
    const float* sX = (const float*)((h & 1) ? xbufB : xbufA);
    if (h + 1 < CH) {
      char* nxt = ((h + 1) & 1) ? xbufB : xbufA;
      stage_x_async(nxt, xbase + (size_t)(h + 1) * CW, tid);
      // 16 newest async ops are h+1's; in-order completion => h's data has landed.
      asm volatile("s_wait_asynccnt 0x10" ::: "memory");
    } else {
      asm volatile("s_wait_asynccnt 0x0" ::: "memory");
    }
    __syncthreads();

    {
      const int wi0 = tid;           // two columns per thread, weights amortized
      const int wi1 = tid + NT;
      float a0[4] = {sSh[0], sSh[1], sSh[2], sSh[3]};
      float a1[4] = {sSh[0], sSh[1], sSh[2], sSh[3]};
      for (int ci = 0; ci < CIN; ++ci) {
        const float* xr = sX + ci * CW;
        const float* wr = sWgt + ci * 5;
#pragma unroll
        for (int kw = 0; kw < 5; ++kw) {
          float w0 = wr[kw], w1 = wr[160 + kw], w2 = wr[320 + kw], w3 = wr[480 + kw];
          int wc0 = wi0 + kw - 2;
          int wc1 = wi1 + kw - 2;
          float xv0 = (wc0 >= 0 && wc0 < CW) ? xr[wc0] : 0.f;
          float xv1 = (wc1 < CW) ? xr[wc1] : 0.f;   // wc1 >= 254 always
          a0[0] += xv0 * w0; a0[1] += xv0 * w1; a0[2] += xv0 * w2; a0[3] += xv0 * w3;
          a1[0] += xv1 * w0; a1[1] += xv1 * w1; a1[2] += xv1 * w2; a1[3] += xv1 * w3;
        }
      }
#pragma unroll
      for (int br = 0; br < 4; ++br) {
        a0[br] = (a0[br] >= 0.f) ? a0[br] : LRELU * a0[br];
        a1[br] = (a1[br] >= 0.f) ? a1[br] : LRELU * a1[br];
      }
      sQp[wi0 * 32 + h] = (_Float16)a0[0];  sQp[wi1 * 32 + h] = (_Float16)a1[0];
      sKp[wi0 * 32 + h] = (_Float16)a0[1];  sKp[wi1 * 32 + h] = (_Float16)a1[1];
      sV [h * CW + wi0] = (_Float16)a0[2];  sV [h * CW + wi1] = (_Float16)a1[2];
      sPE[h * CW + wi0] = a0[3];            sPE[h * CW + wi1] = a1[3];
    }
    __syncthreads();   // protects cur buffer from next iteration's DMA re-fill
  }

  const _Float16 hz = (_Float16)0.f;

  // ---------------- Phase 2: softmax row stats via S^T tiles ----------------
  // S^T[v',w] tile: A = K^T rows v' (K-dim h), B = Q cols w. D-tile: lane = w,
  // VGPR j = v'  -> row stats are IN-LANE reductions (no per-tile cross-lane ops).
  for (int wt = wave; wt < 32; wt += 8) {
    const int wcol = wt * 16 + lane16;
    const _Float16* qr = sQp + wcol * 32 + half * 8;   // zeros come from padding
    v8h q0 = *(const v8h*)qr;
    v8h q1 = *(const v8h*)(qr + 16);
    v16h bq;
#pragma unroll
    for (int e = 0; e < 8; ++e) { bq[e] = q0[e]; bq[e + 8] = q1[e]; }

    float m = -1e30f, l = 0.f;
    for (int ct = 0; ct < 32; ++ct) {
      const int vp = ct * 16 + lane16;
      v8h klo = *(const v8h*)(sKp + vp * 32 + half * 8);
      v16h ak;
#pragma unroll
      for (int e = 0; e < 8; ++e) { ak[e] = klo[e]; ak[e + 8] = hz; }

      v8f s = {};
      s = __builtin_amdgcn_wmma_f32_16x16x32_f16(false, ak, false, bq, (short)0, s, false, false);

      float tmax = s[0];
#pragma unroll
      for (int j = 1; j < 8; ++j) tmax = fmaxf(tmax, s[j]);
      float nm = fmaxf(m, tmax);
      float ts = 0.f;
#pragma unroll
      for (int j = 0; j < 8; ++j) ts += __expf(s[j] - nm);
      l = l * __expf(m - nm) + ts;
      m = nm;
    }
    // merge the two lane halves (v' 8k..8k+7 vs 8k+8..8k+15): one exchange total
    float mo = __shfl_xor(m, 16, 32);
    float lo = __shfl_xor(l, 16, 32);
    float nm = fmaxf(m, mo);
    float lt = l * __expf(m - nm) + lo * __expf(mo - nm);
    if (half == 0) { sM[wcol] = nm; sL[wcol] = 1.0f / lt; }   // store reciprocal
  }
  __syncthreads();

  // ---------------- Phase 3: out = V * softmax(S) + PE ----------------
  _Float16* sPw = sP + wave * (16 * 32);
  const size_t outBase = (size_t)bc * (CH * CW);

  for (int ct = wave; ct < 32; ct += 8) {
    const int vp = ct * 16 + lane16;
    const _Float16* kr = sKp + vp * 32 + half * 8;   // hoisted B-fragment of K
    v8h k0 = *(const v8h*)kr;
    v8h k1 = *(const v8h*)(kr + 16);
    v16h bk;
#pragma unroll
    for (int e = 0; e < 8; ++e) { bk[e] = k0[e]; bk[e + 8] = k1[e]; }

    v8f acc = {};
    for (int rtp = 0; rtp < 16; ++rtp) {        // 32 w-rows per iteration
#pragma unroll
      for (int sub = 0; sub < 2; ++sub) {
        const int rt = rtp * 2 + sub;
        const int wrow = rt * 16 + lane16;
        v8h qlo = *(const v8h*)(sQp + wrow * 32 + half * 8);
        v16h aq;
#pragma unroll
        for (int e = 0; e < 8; ++e) { aq[e] = qlo[e]; aq[e + 8] = hz; }

        v8f s = {};
        s = __builtin_amdgcn_wmma_f32_16x16x32_f16(false, aq, false, bk, (short)0, s, false, false);

        v4f m0 = *(const v4f*)&sM[rt * 16 + half * 8];
        v4f m1 = *(const v4f*)&sM[rt * 16 + half * 8 + 4];
        v4f l0 = *(const v4f*)&sL[rt * 16 + half * 8];
        v4f l1 = *(const v4f*)&sL[rt * 16 + half * 8 + 4];
        v8h ph;
#pragma unroll
        for (int j = 0; j < 8; ++j) {
          float mj = (j < 4) ? m0[j] : m1[j - 4];
          float lj = (j < 4) ? l0[j] : l1[j - 4];
          ph[j] = (_Float16)(__expf(s[j] - mj) * lj);
        }
        // P^T scratch (wave-private): [v'local][wlocal], wlocal = sub*16 + half*8 + j
        *(v8h*)(sPw + lane16 * 32 + sub * 16 + half * 8) = ph;
      }
      // wave-private LDS; DS ops are in-order within a wave -> no barrier needed

      const _Float16* vr = sV + lane16 * CW + rtp * 32 + half * 8;
      v8h v0 = *(const v8h*)vr;
      v8h v1 = *(const v8h*)(vr + 16);
      v16h av;
#pragma unroll
      for (int e = 0; e < 8; ++e) { av[e] = v0[e]; av[e + 8] = v1[e]; }

      const _Float16* pr = sPw + lane16 * 32 + half * 8;
      v8h p0 = *(const v8h*)pr;
      v8h p1 = *(const v8h*)(pr + 16);
      v16h bpv;
#pragma unroll
      for (int e = 0; e < 8; ++e) { bpv[e] = p0[e]; bpv[e + 8] = p1[e]; }

      acc = __builtin_amdgcn_wmma_f32_16x16x32_f16(false, av, false, bpv, (short)0, acc, false, false);
    }
#pragma unroll
    for (int j = 0; j < 8; ++j) {
      int h = half * 8 + j;
      out[outBase + (size_t)h * CW + vp] = acc[j] + sPE[h * CW + vp];
    }
  }
}

extern "C" void kernel_launch(void* const* d_in, const int* in_sizes, int n_in,
                              void* d_out, int out_size, void* d_ws, size_t ws_size,
                              hipStream_t stream) {
  (void)in_sizes; (void)n_in; (void)out_size; (void)d_ws; (void)ws_size;
  const float* x = (const float*)d_in[0];
  const float* W[4]; const float* G[4]; const float* Be[4]; const float* Mu[4]; const float* Va[4];
  for (int br = 0; br < 4; ++br) {
    W [br] = (const float*)d_in[1 + br * 5 + 0];
    G [br] = (const float*)d_in[1 + br * 5 + 1];
    Be[br] = (const float*)d_in[1 + br * 5 + 2];
    Mu[br] = (const float*)d_in[1 + br * 5 + 3];
    Va[br] = (const float*)d_in[1 + br * 5 + 4];
  }
  float* out = (float*)d_out;

  (void)hipFuncSetAttribute((const void*)convattn_kernel,
                            hipFuncAttributeMaxDynamicSharedMemorySize, SM_TOTAL);

  convattn_kernel<<<dim3(CB * COUT), dim3(NT), SM_TOTAL, stream>>>(
      x,
      W[0], G[0], Be[0], Mu[0], Va[0],
      W[1], G[1], Be[1], Mu[1], Va[1],
      W[2], G[2], Be[2], Mu[2], Va[2],
      W[3], G[3], Be[3], Mu[3], Va[3],
      out);
}
